// SoftPoolingDGATEncoder_48747878809865
// MI455X (gfx1250) — matrix operand order repaired and checked by reading, not compiled, not run on hardware
//
#include <hip/hip_runtime.h>
#include <hip/hip_bf16.h>

#define NEG_INF_F (-9.0e15f)
#define LRELU_SLOPE 0.2f

typedef __attribute__((ext_vector_type(16))) __bf16 bf16x16;
typedef __attribute__((ext_vector_type(8)))  __bf16 bf16x8;
typedef __attribute__((ext_vector_type(8)))  float  f32x8;
typedef __attribute__((ext_vector_type(4)))  float  f32x4;

__device__ __forceinline__ f32x8 wmma_bf16(bf16x16 a, bf16x16 b, f32x8 c) {
  // D = A(16x32 bf16) * B(32x16 bf16) + C(16x16 f32)
  return __builtin_amdgcn_wmma_f32_16x16x32_bf16(
      /*neg_a=*/false, a, /*neg_b=*/false, b,
      /*c_mod=*/(short)0, c, /*reuse_a=*/false, /*reuse_b=*/false);
}

// ---------------------------------------------------------------------------
// Per-head GEMM:  Hout[h][n][f] = A[n][k] * W[h][k][f]     (f32 accumulate)
// Also writes transposed bf16 copy HT[h][f][n] for the attention B-matrix.
// One wave (32 threads) computes one 16x16 output tile via WMMA bf16.
// ---------------------------------------------------------------------------
template<bool AF32>
__global__ void gemm_heads_kernel(const void* __restrict__ Aptr,
                                  const float* __restrict__ W,   // [H][K][F]
                                  float* __restrict__ Hout,      // [H][N][F]
                                  __bf16* __restrict__ HT,       // [H][F][N]
                                  int N, int K, int F) {
  const int lane  = threadIdx.x;        // 0..31
  const int r     = lane & 15;
  const int hi    = lane >> 4;          // 0 or 1
  const int kbase = hi ? 8 : 0;
  const int rb    = blockIdx.x * 16;
  const int fb    = blockIdx.y * 16;
  const int h     = blockIdx.z;

  const float* Wh = W + (size_t)h * K * F;
  const int row = rb + r;

  f32x8 acc = {};
  for (int k0 = 0; k0 < K; k0 += 32) {
    // ---- A tile, 16x32, A-register layout:
    // lane(row=r): slots 0..7 <-> K = k0+kbase..+7 ; slots 8..15 <-> K = k0+16+kbase..+7
    bf16x16 a;
    if (AF32) {
      const float* Arow = (const float*)Aptr + (size_t)row * K;
      f32x4 a0 = *(const f32x4*)(Arow + k0 + kbase);
      f32x4 a1 = *(const f32x4*)(Arow + k0 + kbase + 4);
      f32x4 a2 = *(const f32x4*)(Arow + k0 + 16 + kbase);
      f32x4 a3 = *(const f32x4*)(Arow + k0 + 16 + kbase + 4);
#pragma unroll
      for (int j = 0; j < 4; ++j) {
        a[j]      = (__bf16)a0[j];
        a[4 + j]  = (__bf16)a1[j];
        a[8 + j]  = (__bf16)a2[j];
        a[12 + j] = (__bf16)a3[j];
      }
    } else {
      const __bf16* Arow = (const __bf16*)Aptr + (size_t)row * K;
      bf16x8 a0 = *(const bf16x8*)(Arow + k0 + kbase);
      bf16x8 a1 = *(const bf16x8*)(Arow + k0 + 16 + kbase);
#pragma unroll
      for (int j = 0; j < 8; ++j) { a[j] = a0[j]; a[8 + j] = a1[j]; }
    }
    // ---- B tile, 32x16, B-register layout: lane column f = fb+r,
    // slots 0..7 <-> K = k0+kbase..+7 ; slots 8..15 <-> K = k0+16+kbase..+7 (strided f32 loads from W).
    bf16x16 b;
    const int kr = k0 + kbase;
    const float* wcol = Wh + (size_t)kr * F + (fb + r);
#pragma unroll
    for (int j = 0; j < 16; ++j) b[j] = (__bf16)wcol[(size_t)(j < 8 ? j : j + 8) * F];

    acc = wmma_bf16(a, b, acc);
  }

  // ---- Epilogue: C layout: lane col N = r, VGPR v <-> row M = v + 8*hi
  const int moff = hi ? 8 : 0;
  const int fcol = fb + r;
  float* Hh = Hout + ((size_t)h * N + rb + moff) * F + fcol;
  bf16x8 tpack;
#pragma unroll
  for (int v = 0; v < 8; ++v) {
    Hh[(size_t)v * F] = acc[v];
    tpack[v] = (__bf16)acc[v];
  }
  // transposed store: fixed f, 8 consecutive nodes -> one b128
  *(bf16x8*)(HT + ((size_t)h * F + fcol) * N + rb + moff) = tpack;
}

// ---------------------------------------------------------------------------
// Attention coefficient vectors: e_src[h][n] = h[h][n][:].a_src, e_dst likewise
// ---------------------------------------------------------------------------
__global__ void evec_kernel(const float* __restrict__ Hf, const float* __restrict__ A,
                            float* __restrict__ esrc, float* __restrict__ edst,
                            int N, int F) {
  const int idx = blockIdx.x * blockDim.x + threadIdx.x;
  const int h = idx / N, n = idx % N;
  const float* hp = Hf + ((size_t)h * N + n) * F;
  const float* av = A + (size_t)h * 2 * F;
  float s = 0.f, d = 0.f;
  for (int f = 0; f < F; ++f) { float x = hp[f]; s += x * av[f]; d += x * av[F + f]; }
  esrc[h * N + n] = s;
  edst[h * N + n] = d;
}

// ---------------------------------------------------------------------------
// Fused GAT attention (flash-style online softmax + P.V via WMMA).
// One wave per (head, 16-row block). Streams 32-column chunks of adj/e_dst,
// builds the probability tile directly in WMMA A-layout, multiplies against
// hT tiles in B-layout, rescales the f32 accumulator online.
// Epilogue: CONCAT -> elu + bf16 store into [N][H*F]; else f32 [H][N][F].
// ---------------------------------------------------------------------------
template<int F, bool CONCAT>
__global__ void gat_attn_kernel(const float* __restrict__ adj,   // [N][N]
                                const __bf16* __restrict__ HT,   // [H][F][N]
                                const float* __restrict__ esrc,  // [H][N]
                                const float* __restrict__ edst,  // [H][N]
                                const float* __restrict__ acv,   // [H]
                                const float* __restrict__ bcv,   // [H]
                                void* __restrict__ outp,
                                int N, int H) {
  const int lane  = threadIdx.x;
  const int r     = lane & 15;
  const int hi    = lane >> 4;
  const int kbase = hi ? 8 : 0;
  const int rb    = blockIdx.x * 16;
  const int h     = blockIdx.y;
  const int row   = rb + r;
  const int moff  = hi ? 8 : 0;

  const float es = esrc[h * N + row];
  const float ac = acv[h], bc = bcv[h];
  const float* adjrow = adj + (size_t)row * N;
  const float* edh    = edst + (size_t)h * N;
  const __bf16* HTh   = HT + (size_t)h * F * N;

  __shared__ float bro[16];

  constexpr int NT = F / 16;
  f32x8 acc[NT];
#pragma unroll
  for (int t = 0; t < NT; ++t) acc[t] = (f32x8){};

  float rowmax = NEG_INF_F;
  float rowsum = 0.f;

  for (int c0 = 0; c0 < N; c0 += 32) {
    const int ca = c0 + kbase;       // 8 contiguous columns
    const int cb = ca + 16;          // 8 contiguous columns
    f32x4 aj0 = *(const f32x4*)(adjrow + ca);
    f32x4 aj1 = *(const f32x4*)(adjrow + ca + 4);
    f32x4 aj2 = *(const f32x4*)(adjrow + cb);
    f32x4 aj3 = *(const f32x4*)(adjrow + cb + 4);
    f32x4 ed0 = *(const f32x4*)(edh + ca);
    f32x4 ed1 = *(const f32x4*)(edh + ca + 4);
    f32x4 ed2 = *(const f32x4*)(edh + cb);
    f32x4 ed3 = *(const f32x4*)(edh + cb + 4);

    float p[16];
    auto comp = [&](float aval, float ed) -> float {
      float g = ac * aval + bc;
      g = (g > 0.f) ? g : (LRELU_SLOPE * g);
      float et = g * (es + ed);
      return (aval > 0.f) ? et : NEG_INF_F;
    };
#pragma unroll
    for (int j = 0; j < 4; ++j) {
      p[j]      = comp(aj0[j], ed0[j]);
      p[4 + j]  = comp(aj1[j], ed1[j]);
      p[8 + j]  = comp(aj2[j], ed2[j]);
      p[12 + j] = comp(aj3[j], ed3[j]);
    }

    float m = NEG_INF_F;
#pragma unroll
    for (int j = 0; j < 16; ++j) m = fmaxf(m, p[j]);
    m = fmaxf(m, __shfl_xor(m, 16, 32));          // pair-lane (same row) reduce
    const float newmax = fmaxf(rowmax, m);
    const float scale  = __expf(rowmax - newmax);

    float csum = 0.f;
    bf16x16 aP;                                    // probabilities, A-layout
#pragma unroll
    for (int j = 0; j < 16; ++j) {
      float pe = __expf(p[j] - newmax);
      csum += pe;
      aP[j] = (__bf16)pe;
    }
    csum += __shfl_xor(csum, 16, 32);
    rowsum = rowsum * scale + csum;
    rowmax = newmax;

    // broadcast per-row rescale factor to C-layout lanes
    bro[r] = scale;
    __syncthreads();
    float s8[8];
#pragma unroll
    for (int v = 0; v < 8; ++v) s8[v] = bro[moff + v];
    __syncthreads();

    const int kr = c0 + kbase;
#pragma unroll
    for (int t = 0; t < NT; ++t) {
#pragma unroll
      for (int v = 0; v < 8; ++v) acc[t][v] *= s8[v];
      bf16x16 bH;                                  // h tile, B-layout
      const __bf16* hcol = HTh + (size_t)(t * 16 + r) * N + kr;
      bf16x8 b0 = *(const bf16x8*)(hcol);
      bf16x8 b1 = *(const bf16x8*)(hcol + 16);
#pragma unroll
      for (int j = 0; j < 8; ++j) { bH[j] = b0[j]; bH[8 + j] = b1[j]; }
      acc[t] = wmma_bf16(aP, bH, acc[t]);
    }
  }

  // normalize by row sums
  bro[r] = rowsum;
  __syncthreads();
  float inv8[8];
#pragma unroll
  for (int v = 0; v < 8; ++v) inv8[v] = 1.f / bro[moff + v];

  if (CONCAT) {
    __bf16* out = (__bf16*)outp;                   // [N][H*F]
    const int stride = H * F;
#pragma unroll
    for (int t = 0; t < NT; ++t)
#pragma unroll
      for (int v = 0; v < 8; ++v) {
        float x = acc[t][v] * inv8[v];
        x = (x > 0.f) ? x : (__expf(x) - 1.f);     // elu
        out[(size_t)(rb + moff + v) * stride + h * F + t * 16 + r] = (__bf16)x;
      }
  } else {
    float* out = (float*)outp;                     // [H][N][F]
#pragma unroll
    for (int t = 0; t < NT; ++t)
#pragma unroll
      for (int v = 0; v < 8; ++v)
        out[((size_t)h * N + rb + moff + v) * F + t * 16 + r] = acc[t][v] * inv8[v];
  }
}

// ---------------------------------------------------------------------------
// mean over heads -> elu -> max over nodes (one block per feature column)
// ---------------------------------------------------------------------------
__global__ void mean_elu_max_kernel(const float* __restrict__ hp, // [H][N][F]
                                    float* __restrict__ outmax,   // [F]
                                    int N, int F, int H) {
  const int f = blockIdx.x;
  const int tid = threadIdx.x;
  const float invH = 1.f / (float)H;
  float lm = -3.4e38f;
  for (int n = tid; n < N; n += blockDim.x) {
    float s = 0.f;
    for (int hh = 0; hh < H; ++hh) s += hp[((size_t)hh * N + n) * F + f];
    float v = s * invH;
    v = (v > 0.f) ? v : (__expf(v) - 1.f);
    lm = fmaxf(lm, v);
  }
  __shared__ float red[256];
  red[tid] = lm;
  __syncthreads();
  for (int s = 128; s > 0; s >>= 1) {
    if (tid < s) red[tid] = fmaxf(red[tid], red[tid + s]);
    __syncthreads();
  }
  if (tid == 0) outmax[f] = red[0];
}

__global__ void final_map_kernel(const float* __restrict__ om, const float* __restrict__ mw,
                                 const float* __restrict__ mb, float* __restrict__ out, int E) {
  const int j = threadIdx.x;
  float s = mb[j];
  for (int f = 0; f < E; ++f) s += om[f] * mw[f * E + j];
  out[j] = s;
}

// ---------------------------------------------------------------------------
extern "C" void kernel_launch(void* const* d_in, const int* in_sizes, int n_in,
                              void* d_out, int out_size, void* d_ws, size_t ws_size,
                              hipStream_t stream) {
  const float* x   = (const float*)d_in[0];
  const float* adj = (const float*)d_in[1];
  const float* w1  = (const float*)d_in[2];
  const float* a1  = (const float*)d_in[3];
  const float* ac1 = (const float*)d_in[4];
  const float* bc1 = (const float*)d_in[5];
  const float* w2  = (const float*)d_in[6];
  const float* a2  = (const float*)d_in[7];
  const float* ac2 = (const float*)d_in[8];
  const float* bc2 = (const float*)d_in[9];
  const float* mw  = (const float*)d_in[10];
  const float* mb  = (const float*)d_in[11];

  constexpr int N = 4096, D = 128, F1 = 64, H1 = 4, F2 = 128, H2 = 6;
  constexpr int D2 = H1 * F1; // 256

  char* ws = (char*)d_ws;
  size_t off = 0;
  auto carve = [&](size_t bytes) -> void* {
    void* p = ws + off;
    off = (off + bytes + 255) & ~(size_t)255;
    return p;
  };
  float*  h1f  = (float*) carve((size_t)H1 * N * F1 * 4);
  __bf16* h1t  = (__bf16*)carve((size_t)H1 * F1 * N * 2);
  float*  e1s  = (float*) carve((size_t)H1 * N * 4);
  float*  e1d  = (float*) carve((size_t)H1 * N * 4);
  __bf16* out1 = (__bf16*)carve((size_t)N * D2 * 2);
  float*  h2f  = (float*) carve((size_t)H2 * N * F2 * 4);
  __bf16* h2t  = (__bf16*)carve((size_t)H2 * F2 * N * 2);
  float*  e2s  = (float*) carve((size_t)H2 * N * 4);
  float*  e2d  = (float*) carve((size_t)H2 * N * 4);
  float*  hp2  = (float*) carve((size_t)H2 * N * F2 * 4);
  float*  omax = (float*) carve((size_t)F2 * 4);

  // Layer 1
  gemm_heads_kernel<true><<<dim3(N / 16, F1 / 16, H1), 32, 0, stream>>>(x, w1, h1f, h1t, N, D, F1);
  evec_kernel<<<(H1 * N) / 256, 256, 0, stream>>>(h1f, a1, e1s, e1d, N, F1);
  gat_attn_kernel<F1, true><<<dim3(N / 16, H1), 32, 0, stream>>>(adj, h1t, e1s, e1d, ac1, bc1, out1, N, H1);

  // Layer 2
  gemm_heads_kernel<false><<<dim3(N / 16, F2 / 16, H2), 32, 0, stream>>>(out1, w2, h2f, h2t, N, D2, F2);
  evec_kernel<<<(H2 * N) / 256, 256, 0, stream>>>(h2f, a2, e2s, e2d, N, F2);
  gat_attn_kernel<F2, false><<<dim3(N / 16, H2), 32, 0, stream>>>(adj, h2t, e2s, e2d, ac2, bc2, hp2, N, H2);

  // Pool + final projection
  mean_elu_max_kernel<<<F2, 256, 0, stream>>>(hp2, omax, N, F2, H2);
  final_map_kernel<<<1, F2, 0, stream>>>(omax, mw, mb, (float*)d_out, F2);
}
